// LargeReasoningModel_76888504533728
// MI455X (gfx1250) — matrix operand order repaired and checked by reading, not compiled, not run on hardware
//
#include <hip/hip_runtime.h>
#include <hip/hip_bf16.h>
#include <math.h>

// ---------------------------------------------------------------------------
// Model dims (compile-time constants from the reference)
// ---------------------------------------------------------------------------
#define Tn   2048        // B*L tokens
#define Dm   768
#define DHm  3072
#define NHh  12
#define HDh  64
#define NEe  4
#define NLl  4
#define Lseq 1024
#define Vv   32000

typedef __attribute__((ext_vector_type(16))) _Float16 v16h;
typedef __attribute__((ext_vector_type(8)))  _Float16 v8h;
typedef __attribute__((ext_vector_type(8)))  float    v8f;

#define ACT_NONE 0
#define ACT_ELU1 1
#define ACT_GELU 2

// ---------------------------------------------------------------------------
// WMMA GEMM:  C[M,N] = epilogue(A[M,K] @ B[K,N] + bias) (+ residual)
// A,B,C f32 in HBM; f16 staged in LDS via packed b128 stores; f32 accum.
// Block = 256 threads = 8 waves (2 x 4); block tile 64x256, K-step 32.
// Each wave computes a 32x64 patch = 2x4 WMMA accumulators (8 WMMA/K-step).
// Requires M%64==0, N%256==0, K%32==0 (true for all calls below).
// ---------------------------------------------------------------------------
#define BM 64
#define BN 256
#define BK 32
#define LDA_P (BK + 8)    // 40 halves / 80B per A row: 16B aligned, banks spread
#define LDB_P (BN + 8)    // 264 halves / 528B per B row: 16B aligned

static __device__ inline v16h combine16(v8h lo, v8h hi) {
    v16h r;
#pragma unroll
    for (int j = 0; j < 8; ++j) { r[j] = lo[j]; r[j + 8] = hi[j]; }
    return r;
}

__global__ __launch_bounds__(256) void gemm_wmma_kernel(
    const float* __restrict__ A, const float* __restrict__ B,
    const float* __restrict__ bias, const float* __restrict__ res,
    float* __restrict__ C, int M, int N, int K, int act)
{
    __shared__ _Float16 sA[BM * LDA_P];
    __shared__ _Float16 sB[BK * LDB_P];

    const int tid  = threadIdx.x;
    const int lane = tid & 31;
    const int wave = tid >> 5;        // 0..7
    const int wm   = wave >> 2;       // 0..1  (M dir, 32 rows each)
    const int wn   = wave & 3;        // 0..3  (N dir, 64 cols each)
    const int blockM = blockIdx.y * BM;
    const int blockN = blockIdx.x * BN;

    v8f acc[2][4];
#pragma unroll
    for (int i = 0; i < 2; ++i)
#pragma unroll
        for (int j = 0; j < 4; ++j) acc[i][j] = (v8f)(0.0f);

    const int mrow = lane & 15;
    const int kg   = (lane >> 4) * 8;   // ISA A-layout: lanes 16-31 hold K+8

    // Staging geometry (contiguous per-thread runs -> packed b128 DS stores)
    const int ar = tid >> 2;            // A row 0..63
    const int ac = (tid & 3) * 8;       // A col 0,8,16,24
    const int br = tid >> 3;            // B row 0..31
    const int bc = (tid & 7) * 32;      // B col 0,32,...,224

    for (int k0 = 0; k0 < K; k0 += BK) {
        // ---- Stage A tile (64x32): 8 contiguous halves per thread ----
        {
            const float* ga = &A[(size_t)(blockM + ar) * K + k0 + ac];
            float4 f0 = *(const float4*)(ga);
            float4 f1 = *(const float4*)(ga + 4);
            v8h h;
            h[0] = (_Float16)f0.x; h[1] = (_Float16)f0.y;
            h[2] = (_Float16)f0.z; h[3] = (_Float16)f0.w;
            h[4] = (_Float16)f1.x; h[5] = (_Float16)f1.y;
            h[6] = (_Float16)f1.z; h[7] = (_Float16)f1.w;
            *(v8h*)&sA[ar * LDA_P + ac] = h;
        }
        // ---- Stage B tile (32x256): 32 contiguous halves per thread ----
        {
            const float* gb = &B[(size_t)(k0 + br) * N + blockN + bc];
#pragma unroll
            for (int u = 0; u < 4; ++u) {
                float4 f0 = *(const float4*)(gb + u * 8);
                float4 f1 = *(const float4*)(gb + u * 8 + 4);
                v8h h;
                h[0] = (_Float16)f0.x; h[1] = (_Float16)f0.y;
                h[2] = (_Float16)f0.z; h[3] = (_Float16)f0.w;
                h[4] = (_Float16)f1.x; h[5] = (_Float16)f1.y;
                h[6] = (_Float16)f1.z; h[7] = (_Float16)f1.w;
                *(v8h*)&sB[br * LDB_P + bc + u * 8] = h;
            }
        }
        __syncthreads();

        // A fragments: 16x32, lane holds row (lane&15), K in two 16B runs
        v16h afrag[2];
#pragma unroll
        for (int i = 0; i < 2; ++i) {
            const _Float16* ap = &sA[(wm * 32 + i * 16 + mrow) * LDA_P + kg];
            v8h lo = *(const v8h*)(ap);
            v8h hi = *(const v8h*)(ap + 16);
            afrag[i] = combine16(lo, hi);
        }
        // B fragments: 32x16, lane holds col (lane&15), K in same order as A
        v16h bfrag[4];
#pragma unroll
        for (int j = 0; j < 4; ++j) {
            const _Float16* bp = &sB[kg * LDB_P + wn * 64 + j * 16 + mrow];
#pragma unroll
            for (int s = 0; s < 8; ++s) {
                bfrag[j][s]     = bp[s * LDB_P];
                bfrag[j][s + 8] = bp[(s + 16) * LDB_P];
            }
        }
#pragma unroll
        for (int i = 0; i < 2; ++i)
#pragma unroll
            for (int j = 0; j < 4; ++j)
                acc[i][j] = __builtin_amdgcn_wmma_f32_16x16x32_f16(
                    false, afrag[i], false, bfrag[j],
                    (short)0, acc[i][j], false, false);
        __syncthreads();
    }

    // Epilogue: C/D layout -> lane (l&15) = col, row = r + 8*(l>=16)
    const int cl = lane & 15;
    const int rh = (lane >> 4) * 8;
#pragma unroll
    for (int i = 0; i < 2; ++i) {
#pragma unroll
        for (int j = 0; j < 4; ++j) {
            int col = blockN + wn * 64 + j * 16 + cl;
            float bv = bias ? bias[col] : 0.0f;
#pragma unroll
            for (int r = 0; r < 8; ++r) {
                int row = blockM + wm * 32 + i * 16 + rh + r;
                float v = acc[i][j][r] + bv;
                if (act == ACT_ELU1)      v = (v > 0.0f) ? (v + 1.0f) : expf(v);
                else if (act == ACT_GELU) v = 0.5f * v * (1.0f + erff(v * 0.70710678118654752f));
                if (res) v += res[(size_t)row * N + col];
                C[(size_t)row * N + col] = v;
            }
        }
    }
}

// ---------------------------------------------------------------------------
// Embedding gather + positional add: x[t,d] = emb[tok[t],d] + pos[t%L,d]
// ---------------------------------------------------------------------------
__global__ __launch_bounds__(256) void embed_kernel(
    const int* __restrict__ tok, const float* __restrict__ emb,
    const float* __restrict__ pos, float* __restrict__ x)
{
    int i = blockIdx.x * 256 + threadIdx.x;     // over T*D
    int t = i / Dm, d = i - t * Dm;
    int l = t & (Lseq - 1);
    x[i] = emb[(size_t)tok[t] * Dm + d] + pos[(size_t)l * Dm + d];
}

// ---------------------------------------------------------------------------
// LayerNorm: one block per token, 256 threads, biased variance (jnp.var)
// ---------------------------------------------------------------------------
__global__ __launch_bounds__(256) void ln_kernel(
    const float* __restrict__ x, const float* __restrict__ g,
    const float* __restrict__ b, float* __restrict__ y)
{
    __shared__ float red[256];
    const int t = blockIdx.x;
    const float* xr = x + (size_t)t * Dm;
    float s = 0.0f;
    for (int d = threadIdx.x; d < Dm; d += 256) s += xr[d];
    red[threadIdx.x] = s; __syncthreads();
    for (int o = 128; o > 0; o >>= 1) {
        if (threadIdx.x < o) red[threadIdx.x] += red[threadIdx.x + o];
        __syncthreads();
    }
    float mean = red[0] * (1.0f / Dm);
    __syncthreads();
    float v = 0.0f;
    for (int d = threadIdx.x; d < Dm; d += 256) { float z = xr[d] - mean; v += z * z; }
    red[threadIdx.x] = v; __syncthreads();
    for (int o = 128; o > 0; o >>= 1) {
        if (threadIdx.x < o) red[threadIdx.x] += red[threadIdx.x + o];
        __syncthreads();
    }
    float rstd = rsqrtf(red[0] * (1.0f / Dm) + 1e-5f);
    for (int d = threadIdx.x; d < Dm; d += 256)
        y[(size_t)t * Dm + d] = (xr[d] - mean) * rstd * g[d] + b[d];
}

// ---------------------------------------------------------------------------
// Linear-attention global state: KV[b,h,e,f] = sum_l Kf[b,l,h,e]*v[b,l,h,f]
//                                Z[b,h,e]    = sum_l Kf[b,l,h,e]
// One block per (b,h); each thread owns one e-row, 16 f-columns.
// ---------------------------------------------------------------------------
__global__ __launch_bounds__(256) void kv_kernel(
    const float* __restrict__ kf, const float* __restrict__ v,
    float* __restrict__ KV, float* __restrict__ Z)
{
    __shared__ float sk[16 * HDh];
    __shared__ float sv[16 * HDh];
    const int bh = blockIdx.x;                 // b*NH + h
    const int b = bh / NHh, h = bh - b * NHh;
    float acc[16];
#pragma unroll
    for (int i = 0; i < 16; ++i) acc[i] = 0.0f;
    float zacc = 0.0f;
    const int e  = threadIdx.x >> 2;
    const int f0 = (threadIdx.x & 3) * 16;

    for (int l0 = 0; l0 < Lseq; l0 += 16) {
#pragma unroll
        for (int i = 0; i < 4; ++i) {
            int idx = threadIdx.x + i * 256;   // 0..1023
            int lr = idx >> 6, c = idx & 63;
            size_t g = (size_t)(b * Lseq + l0 + lr) * Dm + h * HDh + c;
            sk[lr * HDh + c] = kf[g];
            sv[lr * HDh + c] = v[g];
        }
        __syncthreads();
#pragma unroll
        for (int lr = 0; lr < 16; ++lr) {
            float ke = sk[lr * HDh + e];
#pragma unroll
            for (int i = 0; i < 16; ++i) acc[i] += ke * sv[lr * HDh + f0 + i];
        }
        if (threadIdx.x < HDh) {
#pragma unroll
            for (int lr = 0; lr < 16; ++lr) zacc += sk[lr * HDh + threadIdx.x];
        }
        __syncthreads();
    }
#pragma unroll
    for (int i = 0; i < 16; ++i)
        KV[(size_t)bh * (HDh * HDh) + (size_t)e * HDh + f0 + i] = acc[i];
    if (threadIdx.x < HDh) Z[bh * HDh + threadIdx.x] = zacc;
}

// ---------------------------------------------------------------------------
// out[t,h,f] = (Q[t,h,:] . KV[b,h,:,f]) / (Q[t,h,:] . Z[b,h,:] + 1e-6)
// One block per token.
// ---------------------------------------------------------------------------
__global__ __launch_bounds__(256) void attn_out_kernel(
    const float* __restrict__ q, const float* __restrict__ KV,
    const float* __restrict__ Z, float* __restrict__ ao)
{
    __shared__ float sq[Dm];
    __shared__ float sden[NHh];
    const int t = blockIdx.x;
    const int b = t / Lseq;
    for (int d = threadIdx.x; d < Dm; d += 256) sq[d] = q[(size_t)t * Dm + d];
    __syncthreads();
    if (threadIdx.x < NHh) {
        int h = threadIdx.x;
        float den = 0.0f;
        const float* zp = Z + (b * NHh + h) * HDh;
        for (int e = 0; e < HDh; ++e) den += sq[h * HDh + e] * zp[e];
        sden[h] = den + 1e-6f;
    }
    __syncthreads();
    for (int d = threadIdx.x; d < Dm; d += 256) {
        int h = d >> 6, f = d & 63;
        const float* kvp = KV + (size_t)(b * NHh + h) * (HDh * HDh) + f;
        float o = 0.0f;
        for (int e = 0; e < HDh; ++e) o += sq[h * HDh + e] * kvp[(size_t)e * HDh];
        ao[(size_t)t * Dm + d] = o / sden[h];
    }
}

// ---------------------------------------------------------------------------
// MoE gate: logits = xn @ gw + gb (NE=4); top-2 + softmax. One block/token.
// ---------------------------------------------------------------------------
__global__ __launch_bounds__(256) void gate_kernel(
    const float* __restrict__ xn, const float* __restrict__ gw,
    const float* __restrict__ gb, float* __restrict__ wts, int* __restrict__ idxs)
{
    __shared__ float red[NEe * 256];
    const int t = blockIdx.x;
    const float* xr = xn + (size_t)t * Dm;
    float p[NEe] = {0.f, 0.f, 0.f, 0.f};
    for (int d = threadIdx.x; d < Dm; d += 256) {
        float xv = xr[d];
#pragma unroll
        for (int e = 0; e < NEe; ++e) p[e] += xv * gw[d * NEe + e];
    }
#pragma unroll
    for (int e = 0; e < NEe; ++e) red[e * 256 + threadIdx.x] = p[e];
    __syncthreads();
    for (int o = 128; o > 0; o >>= 1) {
        if (threadIdx.x < o)
#pragma unroll
            for (int e = 0; e < NEe; ++e)
                red[e * 256 + threadIdx.x] += red[e * 256 + threadIdx.x + o];
        __syncthreads();
    }
    if (threadIdx.x == 0) {
        float gl[NEe];
#pragma unroll
        for (int e = 0; e < NEe; ++e) gl[e] = red[e * 256] + gb[e];
        int i0 = 0;
        for (int e = 1; e < NEe; ++e) if (gl[e] > gl[i0]) i0 = e;   // ties -> first
        int i1 = -1;
        for (int e = 0; e < NEe; ++e) {
            if (e == i0) continue;
            if (i1 < 0 || gl[e] > gl[i1]) i1 = e;
        }
        float e0 = expf(gl[i0] - gl[i0]);
        float e1 = expf(gl[i1] - gl[i0]);
        float inv = 1.0f / (e0 + e1);
        wts[t * 2 + 0] = e0 * inv;  wts[t * 2 + 1] = e1 * inv;
        idxs[t * 2 + 0] = i0;       idxs[t * 2 + 1] = i1;
    }
}

// ---------------------------------------------------------------------------
// MoE combine: x[t,d] += w0*eo[idx0][t,d] + w1*eo[idx1][t,d]
// ---------------------------------------------------------------------------
__global__ __launch_bounds__(256) void moe_combine_kernel(
    const float* __restrict__ eo, const float* __restrict__ wts,
    const int* __restrict__ idxs, float* __restrict__ x)
{
    int i = blockIdx.x * 256 + threadIdx.x;   // over T*D
    int t = i / Dm;
    float a = x[i];
    a += wts[t * 2 + 0] * eo[(size_t)idxs[t * 2 + 0] * ((size_t)Tn * Dm) + i];
    a += wts[t * 2 + 1] * eo[(size_t)idxs[t * 2 + 1] * ((size_t)Tn * Dm) + i];
    x[i] = a;
}

// ---------------------------------------------------------------------------
// Host orchestration
// ---------------------------------------------------------------------------
extern "C" void kernel_launch(void* const* d_in, const int* in_sizes, int n_in,
                              void* d_out, int out_size, void* d_ws, size_t ws_size,
                              hipStream_t stream)
{
    (void)in_sizes; (void)n_in; (void)out_size; (void)ws_size;
    const int*   tok  = (const int*)d_in[0];
    const float* emb  = (const float*)d_in[1];
    const float* pos  = (const float*)d_in[2];
    const float* qw   = (const float*)d_in[3];   const float* qb = (const float*)d_in[4];
    const float* kw   = (const float*)d_in[5];   const float* kb = (const float*)d_in[6];
    const float* vw   = (const float*)d_in[7];   const float* vb = (const float*)d_in[8];
    const float* ow   = (const float*)d_in[9];   const float* ob = (const float*)d_in[10];
    const float* ln1g = (const float*)d_in[11];  const float* ln1b = (const float*)d_in[12];
    const float* ln2g = (const float*)d_in[13];  const float* ln2b = (const float*)d_in[14];
    const float* gw   = (const float*)d_in[15];  const float* gb = (const float*)d_in[16];
    const float* w1   = (const float*)d_in[17];  const float* b1 = (const float*)d_in[18];
    const float* w2   = (const float*)d_in[19];  const float* b2 = (const float*)d_in[20];
    const float* hw   = (const float*)d_in[21];  const float* hb = (const float*)d_in[22];
    float* out = (float*)d_out;

    // Workspace carve-up (f32 elements)
    const size_t TD  = (size_t)Tn * Dm;       // 1,572,864
    const size_t TDH = (size_t)Tn * DHm;      // 6,291,456
    float* ws   = (float*)d_ws;
    float* x    = ws;                 ws += TD;
    float* xn   = ws;                 ws += TD;
    float* q    = ws;                 ws += TD;
    float* kf   = ws;                 ws += TD;
    float* vbuf = ws;                 ws += TD;
    float* ao   = ws;                 ws += TD;
    float* hbuf = ws;                 ws += TDH;
    float* eo   = ws;                 ws += (size_t)NEe * TD;
    float* KV   = ws;                 ws += (size_t)(2 * NHh) * HDh * HDh;
    float* Z    = ws;                 ws += (size_t)(2 * NHh) * HDh;
    float* wts  = ws;                 ws += (size_t)Tn * 2;
    int*   idxs = (int*)ws;           ws += (size_t)Tn * 2;

    const dim3 blk(256);
    const dim3 gD (Dm  / BN, Tn / BM);   // N=768   -> (3, 32)
    const dim3 gDH(DHm / BN, Tn / BM);   // N=3072  -> (12, 32)
    const dim3 gV (Vv  / BN, Tn / BM);   // N=32000 -> (125, 32)

    embed_kernel<<<(int)(TD / 256), blk, 0, stream>>>(tok, emb, pos, x);

    for (int l = 0; l < NLl; ++l) {
        const size_t wDD = (size_t)l * Dm * Dm;
        // ---- attention ----
        ln_kernel<<<Tn, blk, 0, stream>>>(x, ln1g + l * Dm, ln1b + l * Dm, xn);
        gemm_wmma_kernel<<<gD, blk, 0, stream>>>(xn, qw + wDD, qb + l * Dm, nullptr, q,    Tn, Dm, Dm, ACT_ELU1);
        gemm_wmma_kernel<<<gD, blk, 0, stream>>>(xn, kw + wDD, kb + l * Dm, nullptr, kf,   Tn, Dm, Dm, ACT_ELU1);
        gemm_wmma_kernel<<<gD, blk, 0, stream>>>(xn, vw + wDD, vb + l * Dm, nullptr, vbuf, Tn, Dm, Dm, ACT_NONE);
        kv_kernel<<<2 * NHh, blk, 0, stream>>>(kf, vbuf, KV, Z);
        attn_out_kernel<<<Tn, blk, 0, stream>>>(q, KV, Z, ao);
        gemm_wmma_kernel<<<gD, blk, 0, stream>>>(ao, ow + wDD, ob + l * Dm, x, x, Tn, Dm, Dm, ACT_NONE);
        // ---- MoE ----
        ln_kernel<<<Tn, blk, 0, stream>>>(x, ln2g + l * Dm, ln2b + l * Dm, xn);
        gate_kernel<<<Tn, blk, 0, stream>>>(xn, gw + (size_t)l * Dm * NEe, gb + l * NEe, wts, idxs);
        for (int e = 0; e < NEe; ++e) {
            const size_t le = (size_t)(l * NEe + e);
            gemm_wmma_kernel<<<gDH, blk, 0, stream>>>(
                xn, w1 + le * Dm * DHm, b1 + le * DHm, nullptr, hbuf, Tn, DHm, Dm, ACT_GELU);
            gemm_wmma_kernel<<<gD, blk, 0, stream>>>(
                hbuf, w2 + le * DHm * Dm, b2 + le * Dm, nullptr, eo + (size_t)e * TD, Tn, Dm, DHm, ACT_NONE);
        }
        moe_combine_kernel<<<(int)(TD / 256), blk, 0, stream>>>(eo, wts, idxs, x);
    }

    // ---- LM head ----
    gemm_wmma_kernel<<<gV, blk, 0, stream>>>(x, hw, hb, nullptr, out, Tn, Vv, Dm, ACT_NONE);
}
